// ROIHead_31198642438590
// MI455X (gfx1250) — compile-verified, matrix operand order checked by reading, not run on hardware
//
#include <hip/hip_runtime.h>
#include <hip/hip_bf16.h>

typedef _Float16 v8h  __attribute__((ext_vector_type(8)));
typedef _Float16 v16h __attribute__((ext_vector_type(16)));
typedef float    v8f  __attribute__((ext_vector_type(8)));
typedef int      v4i  __attribute__((ext_vector_type(4)));

#define N_PROP   1024
#define KDIM     25088      // 512*49
#define FCD      1024
#define NCAND    20480      // 1024*20
#define PRE_NMS  2000
#define POST_NMS 100
#define NSORT    32768
#define FH       50
#define FW       50

// GEMM tiling
#define BK         64                       // k-halves per staged tile
#define LDS_STRIDE 72                       // 64 + 8 pad halves -> conflict-free ds_load_b128
#define A_HALFS    (64 * LDS_STRIDE)        // 64-row A tile
#define B_HALFS    (128 * LDS_STRIDE)       // 128-row B tile
#define BUF_HALFS  (A_HALFS + B_HALFS)

// ---------------------------------------------------------------- roi_pool
__global__ __launch_bounds__(256, 1)
void roi_pool_f16_kernel(const float* __restrict__ feat,
                         const float* __restrict__ props,
                         _Float16* __restrict__ A16) {
    int n = blockIdx.x;
    const float* b = props + n * 4;
    float sw = rintf(b[0] * 0.0625f), sh = rintf(b[1] * 0.0625f);
    float ew = rintf(b[2] * 0.0625f), eh = rintf(b[3] * 0.0625f);
    float bw = fmaxf(ew - sw + 1.0f, 1.0f) / 7.0f;
    float bh = fmaxf(eh - sh + 1.0f, 1.0f) / 7.0f;
    for (int idx = threadIdx.x; idx < KDIM; idx += 256) {
        int c = idx / 49;
        int r = idx - c * 49;
        int ph = r / 7, pw = r - ph * 7;
        float hsf = fminf(fmaxf(floorf((float)ph * bh) + sh, 0.0f), 50.0f);
        float hef = fminf(fmaxf(ceilf((float)(ph + 1) * bh) + sh, 0.0f), 50.0f);
        float wsf = fminf(fmaxf(floorf((float)pw * bw) + sw, 0.0f), 50.0f);
        float wef = fminf(fmaxf(ceilf((float)(pw + 1) * bw) + sw, 0.0f), 50.0f);
        int hs = (int)hsf, he = (int)hef, ws = (int)wsf, we = (int)wef;
        float m = -__builtin_inff();
        const float* fc = feat + (size_t)c * (FH * FW);
        for (int h = hs; h < he; ++h)
            for (int w = ws; w < we; ++w)
                m = fmaxf(m, fc[h * FW + w]);
        if (!(m >= -3.0e38f)) m = 0.0f;   // empty bin -> 0
        A16[(size_t)n * KDIM + idx] = (_Float16)m;
    }
}

// ------------------------------------------- transpose + f32 -> f16 convert
__global__ __launch_bounds__(256, 1)
void transpose_cvt_kernel(const float* __restrict__ in, _Float16* __restrict__ out,
                          int K, int N) {
    // in: [K,N] row-major f32  ->  out: [N,K] row-major f16
    __shared__ float tile[32][33];
    int k0 = blockIdx.x * 32, n0 = blockIdx.y * 32;
    int tx = threadIdx.x & 31, ty = threadIdx.x >> 5;  // 32 x 8
    for (int r = ty; r < 32; r += 8) {
        int k = k0 + r, n = n0 + tx;
        tile[r][tx] = (k < K && n < N) ? in[(size_t)k * N + n] : 0.0f;
    }
    __syncthreads();
    for (int r = ty; r < 32; r += 8) {
        int n = n0 + r, k = k0 + tx;
        if (n < N && k < K) out[(size_t)n * K + k] = (_Float16)tile[tx][r];
    }
}

// ---------------------------------------------------------------- WMMA GEMM
// Fragment layout (ISA 7.12.2, 16-bit A 16x32): lane<16 holds K 0..7 & 16..23,
// lane>=16 holds K 8..15 & 24..31, M = lane&15. B fragments use the same
// pattern on the N-major (transposed) weight matrix.
__device__ __forceinline__
v16h lds_frag(const _Float16* __restrict__ base, int r0, int kk, int lane) {
    int m = lane & 15, hi = lane >> 4;
    const _Float16* p = base + (r0 + m) * LDS_STRIDE + kk + (hi << 3);
    v8h lo = *reinterpret_cast<const v8h*>(p);
    v8h hh = *reinterpret_cast<const v8h*>(p + 16);
    v16h f;
#pragma unroll
    for (int i = 0; i < 8; ++i) { f[i] = lo[i]; f[8 + i] = hh[i]; }
    return f;
}

__device__ __forceinline__
void store_ctile(v8f acc, _Float16* __restrict__ C, const float* __restrict__ bias,
                 int N, int r0, int c0, int lane, int relu) {
    int n  = c0 + (lane & 15);
    int rb = r0 + ((lane >> 4) << 3);
    float bv = bias[n];
#pragma unroll
    for (int i = 0; i < 8; ++i) {
        float v = acc[i] + bv;
        if (relu) v = fmaxf(v, 0.0f);
        C[(size_t)(rb + i) * N + n] = (_Float16)v;
    }
}

// Block tile 64(M) x 128(N) x 64(K); 8 waves, each 32x32; LDS double-buffered.
__global__ __launch_bounds__(256, 1)
void gemm_wmma_kernel(const _Float16* __restrict__ A,   // [M,K] f16
                      const _Float16* __restrict__ Bt,  // [N,K] f16
                      const float* __restrict__ bias,   // [N]
                      _Float16* __restrict__ C,         // [M,N] f16
                      int M, int N, int K, int relu) {
    __shared__ __align__(16) _Float16 smem[2 * BUF_HALFS];   // 55,296 B (CDNA5 LDS)

    int tid  = threadIdx.x;
    int lane = tid & 31;
    int wid  = tid >> 5;
    int waveN = wid & 3, waveM = wid >> 2;              // 2 x 4 waves
    int rowB0 = blockIdx.y * 64;
    int colB0 = blockIdx.x * 128;
    int wr = waveM * 32;                                // wave row within A tile
    int wc = waveN * 32;                                // wave col within B tile

    const int nk = K / BK;

    // staging registers: A = 2 chunks, B = 4 chunks of 16B per thread
    v4i rA[2], rB[4];

    auto gload = [&](int k0) {
#pragma unroll
        for (int i = 0; i < 2; ++i) {
            int cid = tid + 256 * i;
            int r = cid >> 3, q = cid & 7;
            rA[i] = *reinterpret_cast<const v4i*>(A + (size_t)(rowB0 + r) * K + k0 + q * 8);
        }
#pragma unroll
        for (int i = 0; i < 4; ++i) {
            int cid = tid + 256 * i;
            int r = cid >> 3, q = cid & 7;
            rB[i] = *reinterpret_cast<const v4i*>(Bt + (size_t)(colB0 + r) * K + k0 + q * 8);
        }
    };
    auto sstore = [&](int buf) {
        _Float16* sA = smem + buf * BUF_HALFS;
        _Float16* sB = sA + A_HALFS;
#pragma unroll
        for (int i = 0; i < 2; ++i) {
            int cid = tid + 256 * i;
            int r = cid >> 3, q = cid & 7;
            *reinterpret_cast<v4i*>(sA + r * LDS_STRIDE + q * 8) = rA[i];
        }
#pragma unroll
        for (int i = 0; i < 4; ++i) {
            int cid = tid + 256 * i;
            int r = cid >> 3, q = cid & 7;
            *reinterpret_cast<v4i*>(sB + r * LDS_STRIDE + q * 8) = rB[i];
        }
    };

    v8f acc00 = {}, acc01 = {}, acc10 = {}, acc11 = {};

    gload(0);
    sstore(0);
    __syncthreads();

    for (int kt = 0; kt < nk; ++kt) {
        int cur = kt & 1;
        if (kt + 1 < nk) gload((kt + 1) * BK);          // overlap with compute below
        if (kt + 2 < nk) {                              // advisory L2 prefetch (global_prefetch_b8)
            int k2 = (kt + 2) * BK;
            const _Float16* pp = (tid < 64)
                ? (A  + (size_t)(rowB0 + tid) * K + k2)
                : (Bt + (size_t)(colB0 + ((tid - 64) & 127)) * K + k2);
            __builtin_prefetch(pp, 0, 1);
        }

        const _Float16* sA = smem + cur * BUF_HALFS;
        const _Float16* sB = sA + A_HALFS;
#pragma unroll
        for (int kk = 0; kk < BK; kk += 32) {
            v16h a0 = lds_frag(sA, wr,      kk, lane);
            v16h a1 = lds_frag(sA, wr + 16, kk, lane);
            v16h b0 = lds_frag(sB, wc,      kk, lane);
            v16h b1 = lds_frag(sB, wc + 16, kk, lane);
            acc00 = __builtin_amdgcn_wmma_f32_16x16x32_f16(false, a0, false, b0, (short)0, acc00, false, false);
            acc01 = __builtin_amdgcn_wmma_f32_16x16x32_f16(false, a0, false, b1, (short)0, acc01, false, false);
            acc10 = __builtin_amdgcn_wmma_f32_16x16x32_f16(false, a1, false, b0, (short)0, acc10, false, false);
            acc11 = __builtin_amdgcn_wmma_f32_16x16x32_f16(false, a1, false, b1, (short)0, acc11, false, false);
        }
        if (kt + 1 < nk) sstore(1 - cur);
        __syncthreads();
    }

    int row0 = rowB0 + wr, col0 = colB0 + wc;
    store_ctile(acc00, C, bias, N, row0,      col0,      lane, relu);
    store_ctile(acc01, C, bias, N, row0,      col0 + 16, lane, relu);
    store_ctile(acc10, C, bias, N, row0 + 16, col0,      lane, relu);
    store_ctile(acc11, C, bias, N, row0 + 16, col0 + 16, lane, relu);
}

// ------------------------------------- cls/reg heads + softmax + box decode
__global__ __launch_bounds__(128, 1)
void head_kernel(const _Float16* __restrict__ X,       // [1024,1024] f16
                 const float* __restrict__ props,
                 const float* __restrict__ Wc, const float* __restrict__ bc,
                 const float* __restrict__ Wr, const float* __restrict__ br,
                 const int* __restrict__ pih, const int* __restrict__ piw,
                 float* __restrict__ cbox, float* __restrict__ cscore,
                 int* __restrict__ clabel, int* __restrict__ cvalid) {
    __shared__ float x[FCD];
    __shared__ float cls[21];
    __shared__ float reg[84];
    __shared__ float probs[21];
    int n = blockIdx.x, t = threadIdx.x;
    for (int i = t; i < FCD; i += 128) x[i] = (float)X[(size_t)n * FCD + i];
    __syncthreads();
    if (t < 105) {
        if (t < 21) {
            float s = bc[t];
            for (int k = 0; k < FCD; ++k) s += x[k] * Wc[k * 21 + t];
            cls[t] = s;
        } else {
            int j = t - 21;
            float s = br[j];
            for (int k = 0; k < FCD; ++k) s += x[k] * Wr[k * 84 + j];
            reg[j] = s;
        }
    }
    __syncthreads();
    if (t == 0) {
        float mx = cls[0];
        for (int i = 1; i < 21; ++i) mx = fmaxf(mx, cls[i]);
        float sum = 0.0f;
        for (int i = 0; i < 21; ++i) { probs[i] = __expf(cls[i] - mx); sum += probs[i]; }
        float inv = 1.0f / sum;
        for (int i = 0; i < 21; ++i) probs[i] *= inv;
    }
    __syncthreads();
    if (t < 20) {
        const float BBOX_CLIP = 4.1351666f;   // log(1000/16)
        int c = t + 1;
        const float* pb = props + n * 4;
        float w = pb[2] - pb[0], h = pb[3] - pb[1];
        float cx = pb[0] + 0.5f * w, cy = pb[1] + 0.5f * h;
        float dx = reg[c * 4 + 0], dy = reg[c * 4 + 1];
        float dw = fminf(reg[c * 4 + 2], BBOX_CLIP);
        float dh = fminf(reg[c * 4 + 3], BBOX_CLIP);
        float pcx = dx * w + cx, pcy = dy * h + cy;
        float pw2 = __expf(dw) * w, ph2 = __expf(dh) * h;
        float imw = (float)piw[0], imh = (float)pih[0];
        float x1 = fminf(fmaxf(pcx - 0.5f * pw2, 0.0f), imw);
        float y1 = fminf(fmaxf(pcy - 0.5f * ph2, 0.0f), imh);
        float x2 = fminf(fmaxf(pcx + 0.5f * pw2, 0.0f), imw);
        float y2 = fminf(fmaxf(pcy + 0.5f * ph2, 0.0f), imh);
        float sc = probs[c];
        int valid = (sc > 0.05f) && ((x2 - x1) >= 1.0f) && ((y2 - y1) >= 1.0f);
        int o = n * 20 + t;
        cbox[o * 4 + 0] = x1; cbox[o * 4 + 1] = y1;
        cbox[o * 4 + 2] = x2; cbox[o * 4 + 3] = y2;
        cscore[o] = sc; clabel[o] = c; cvalid[o] = valid;
    }
}

// ------------------- single-WG bitonic sort of 32768 keys in 256 KB of LDS
__global__ __launch_bounds__(1024, 1)
void sort_kernel(const float* __restrict__ cscore, const int* __restrict__ cvalid,
                 int* __restrict__ sidx, float* __restrict__ skey) {
    extern __shared__ char smem[];
    float* key = (float*)smem;
    int*   val = (int*)(smem + (size_t)NSORT * sizeof(float));
    int tid = threadIdx.x;
    for (int i = tid; i < NSORT; i += 1024) {
        float kf = -3.0e38f;
        if (i < NCAND) kf = cvalid[i] ? cscore[i] : -1.0f;   // masked score
        key[i] = kf; val[i] = i;
    }
    __syncthreads();
    for (int k = 2; k <= NSORT; k <<= 1) {
        for (int j = k >> 1; j > 0; j >>= 1) {
            for (int i = tid; i < NSORT; i += 1024) {
                int ixj = i ^ j;
                if (ixj > i) {
                    bool up = ((i & k) == 0);
                    float ki = key[i], kj = key[ixj];
                    bool sw = up ? (ki < kj) : (ki > kj);   // descending
                    if (sw) {
                        key[i] = kj; key[ixj] = ki;
                        int tv = val[i]; val[i] = val[ixj]; val[ixj] = tv;
                    }
                }
            }
            __syncthreads();
        }
    }
    for (int i = tid; i < PRE_NMS; i += 1024) { sidx[i] = val[i]; skey[i] = key[i]; }
}

// ------------------------------------------------ greedy class-aware NMS
__global__ __launch_bounds__(1024, 1)
void nms_kernel(const float* __restrict__ cbox, const int* __restrict__ clabel,
                const int* __restrict__ cvalid, const int* __restrict__ sidx,
                int* __restrict__ keepg) {
    __shared__ float bx1[PRE_NMS], by1[PRE_NMS], bx2[PRE_NMS], by2[PRE_NMS], ar[PRE_NMS];
    __shared__ int lab[PRE_NMS];
    __shared__ int keep[PRE_NMS];
    int tid = threadIdx.x;
    for (int j = tid; j < PRE_NMS; j += 1024) {
        int g = sidx[j];
        float x1 = cbox[g * 4], y1 = cbox[g * 4 + 1];
        float x2 = cbox[g * 4 + 2], y2 = cbox[g * 4 + 3];
        bx1[j] = x1; by1[j] = y1; bx2[j] = x2; by2[j] = y2;
        ar[j] = (x2 - x1) * (y2 - y1);
        lab[j] = clabel[g]; keep[j] = cvalid[g];
    }
    __syncthreads();
    for (int i = 0; i < PRE_NMS; ++i) {
        if (keep[i]) {   // uniform across block
            float x1 = bx1[i], y1 = by1[i], x2 = bx2[i], y2 = by2[i], ai = ar[i];
            int li = lab[i];
            for (int j = i + 1 + tid; j < PRE_NMS; j += 1024) {
                if (keep[j] && lab[j] == li) {   // class offset => cross-class IoU == 0
                    float ix = fmaxf(0.0f, fminf(x2, bx2[j]) - fmaxf(x1, bx1[j]));
                    float iy = fmaxf(0.0f, fminf(y2, by2[j]) - fmaxf(y1, by1[j]));
                    float inter = ix * iy;
                    float iou = inter / (ai + ar[j] - inter + 1e-9f);
                    if (iou > 0.5f) keep[j] = 0;
                }
            }
        }
        __syncthreads();
    }
    for (int j = tid; j < PRE_NMS; j += 1024) keepg[j] = keep[j];
}

// ------------------------------- compact top-100 survivors into the output
__global__ void final_kernel(const float* __restrict__ cbox, const float* __restrict__ cscore,
                             const int* __restrict__ clabel, const int* __restrict__ sidx,
                             const int* __restrict__ keep, float* __restrict__ out) {
    if (threadIdx.x == 0 && blockIdx.x == 0) {
        int cnt = 0;
        for (int j = 0; j < PRE_NMS && cnt < POST_NMS; ++j) {
            if (keep[j]) {
                int g = sidx[j];
                out[cnt * 4 + 0] = cbox[g * 4 + 0];
                out[cnt * 4 + 1] = cbox[g * 4 + 1];
                out[cnt * 4 + 2] = cbox[g * 4 + 2];
                out[cnt * 4 + 3] = cbox[g * 4 + 3];
                out[4 * POST_NMS + cnt] = cscore[g];
                out[5 * POST_NMS + cnt] = (float)clabel[g];
                ++cnt;
            }
        }
        for (; cnt < POST_NMS; ++cnt) {
            out[cnt * 4 + 0] = 0.0f; out[cnt * 4 + 1] = 0.0f;
            out[cnt * 4 + 2] = 0.0f; out[cnt * 4 + 3] = 0.0f;
            out[4 * POST_NMS + cnt] = 0.0f;
            out[5 * POST_NMS + cnt] = 0.0f;
        }
    }
}

// ---------------------------------------------------------------- launcher
extern "C" void kernel_launch(void* const* d_in, const int* in_sizes, int n_in,
                              void* d_out, int out_size, void* d_ws, size_t ws_size,
                              hipStream_t stream) {
    const float* feat  = (const float*)d_in[0];
    const float* props = (const float*)d_in[1];
    const float* W6    = (const float*)d_in[2];
    const float* b6    = (const float*)d_in[3];
    const float* W7    = (const float*)d_in[4];
    const float* b7    = (const float*)d_in[5];
    const float* Wc    = (const float*)d_in[6];
    const float* bc    = (const float*)d_in[7];
    const float* Wr    = (const float*)d_in[8];
    const float* br    = (const float*)d_in[9];
    const int*   pih   = (const int*)d_in[10];
    const int*   piw   = (const int*)d_in[11];

    char* ws = (char*)d_ws;
    size_t off = 0;
    auto carve = [&](size_t bytes) -> char* {
        char* p = ws + off;
        off = (off + bytes + 255) & ~(size_t)255;
        return p;
    };
    _Float16* A16 = (_Float16*)carve((size_t)N_PROP * KDIM * 2);
    _Float16* W6t = (_Float16*)carve((size_t)KDIM * FCD * 2);
    _Float16* X6  = (_Float16*)carve((size_t)N_PROP * FCD * 2);
    _Float16* W7t = (_Float16*)carve((size_t)FCD * FCD * 2);
    _Float16* X7  = (_Float16*)carve((size_t)N_PROP * FCD * 2);
    float* cbox   = (float*)carve((size_t)NCAND * 4 * 4);
    float* cscore = (float*)carve((size_t)NCAND * 4);
    int*   clabel = (int*)carve((size_t)NCAND * 4);
    int*   cvalid = (int*)carve((size_t)NCAND * 4);
    int*   sidx   = (int*)carve((size_t)PRE_NMS * 4);
    float* skey   = (float*)carve((size_t)PRE_NMS * 4);
    int*   keep   = (int*)carve((size_t)PRE_NMS * 4);

    // 1) roi_pool -> f16 activations [1024, 25088]
    roi_pool_f16_kernel<<<N_PROP, 256, 0, stream>>>(feat, props, A16);

    // 2) weight transpose+convert: W6 [25088,1024] -> W6t [1024,25088] f16; same for W7
    transpose_cvt_kernel<<<dim3(KDIM / 32, FCD / 32), 256, 0, stream>>>(W6, W6t, KDIM, FCD);
    transpose_cvt_kernel<<<dim3(FCD / 32, FCD / 32), 256, 0, stream>>>(W7, W7t, FCD, FCD);

    // 3) FC6 (K=25088) and FC7 (K=1024): LDS double-buffered WMMA, fused bias+ReLU
    gemm_wmma_kernel<<<dim3(FCD / 128, N_PROP / 64), 256, 0, stream>>>(A16, W6t, b6, X6, N_PROP, FCD, KDIM, 1);
    gemm_wmma_kernel<<<dim3(FCD / 128, N_PROP / 64), 256, 0, stream>>>(X6, W7t, b7, X7, N_PROP, FCD, FCD, 1);

    // 4) cls/reg heads + softmax + decode -> 20480 candidates
    head_kernel<<<N_PROP, 128, 0, stream>>>(X7, props, Wc, bc, Wr, br, pih, piw,
                                            cbox, cscore, clabel, cvalid);

    // 5) descending bitonic sort of masked scores (256 KB LDS, CDNA5 only)
    sort_kernel<<<1, 1024, (size_t)NSORT * 8, stream>>>(cscore, cvalid, sidx, skey);

    // 6) greedy class-aware NMS over top-2000
    nms_kernel<<<1, 1024, 0, stream>>>(cbox, clabel, cvalid, sidx, keep);

    // 7) top-100 survivors -> d_out (boxes[100,4] | scores[100] | labels[100])
    final_kernel<<<1, 64, 0, stream>>>(cbox, cscore, clabel, sidx, keep, (float*)d_out);
}
